// GNNlayer_19378892439976
// MI455X (gfx1250) — matrix operand order repaired and checked by reading, not compiled, not run on hardware
//
#include <hip/hip_runtime.h>
#include <math.h>

typedef __attribute__((ext_vector_type(16))) _Float16 v16h;
typedef __attribute__((ext_vector_type(8)))  float    v8f;
typedef __attribute__((ext_vector_type(4)))  unsigned int u32x4;
typedef __attribute__((ext_vector_type(8)))  int      i32x8;
typedef __attribute__((ext_vector_type(4)))  int      i32x4;

#define N_NODES 1024
#define DIM     128
#define HEADS   4

// ---------------------------------------------------------------------------
// One 16x16 f32 output tile: C += A(16xK) * B(Kx16), operands converted f32->f16
// on the fly, accumulated in f32 via v_wmma_f32_16x16x32_f16.
//   A(m,k) = Ap[m*sa_m + k*sa_k]
//   B(k,n) = Bp[k*sb_k + n*sb_n]
// Layouts per CDNA5 ISA 7.12.2 (wave32):
//   A 16x32 f16 : lane m=l&15; elems 0..7 <-> K = kk + (l<16?0:8) + e
//                              elems 8..15 <-> K = kk + 16 + (l<16?0:8) + e
//   B 32x16 f16 : lane n=l&15; same K mapping as A (elems 0..7 / 8..15)
//   C/D 16x16 f32: elem v <-> row m = v + (l<16?0:8), col n = l&15
// ---------------------------------------------------------------------------
__device__ __forceinline__ v8f wmma_tile_f16(
    const float* __restrict__ Ap, int sa_m, int sa_k,
    const float* __restrict__ Bp, int sb_k, int sb_n,
    int K, int lane)
{
  v8f c = {0.f, 0.f, 0.f, 0.f, 0.f, 0.f, 0.f, 0.f};
  const int hl  = lane >> 4;     // 0 or 1
  const int idx = lane & 15;     // M for A, N for B
  const int akb = hl * 8;
  const int bkb = hl * 8;
  for (int kk = 0; kk < K; kk += 32) {
    v16h a, b;
#pragma unroll
    for (int e = 0; e < 8; ++e) {
      a[e]     = (_Float16)Ap[idx * sa_m + (kk + akb + e) * sa_k];
      a[e + 8] = (_Float16)Ap[idx * sa_m + (kk + 16 + akb + e) * sa_k];
    }
#pragma unroll
    for (int e = 0; e < 16; ++e) {
      b[e] = (_Float16)Bp[(kk + bkb + e + (e & 8)) * sb_k + idx * sb_n];
    }
    // (neg_a, A, neg_b, B, c_mod, C, reuse_a, reuse_b)
    c = __builtin_amdgcn_wmma_f32_16x16x32_f16(false, a, false, b, (short)0, c,
                                               false, false);
  }
  return c;
}

template <bool LRELU>
__device__ __forceinline__ void store_tile(float* __restrict__ Cp, int ldc,
                                           int lane, v8f c)
{
  const int hl = lane >> 4;
  const int n  = lane & 15;
#pragma unroll
  for (int v = 0; v < 8; ++v) {
    int   m   = v + hl * 8;
    float val = c[v];
    if (LRELU) val = fmaxf(val, 0.01f * val);
    Cp[m * ldc + n] = val;
  }
}

// ---------------------------------------------------------------------------
// TDM: DMA a 2-D tile (tile_d1 rows x tile_d0 elems, f32) from a row-major
// tensor (row stride d0_stride elems) in global memory into LDS, packed
// row-major at lds_byte_addr.  D# packing per CDNA5 ISA ch.8.
// Issued by a single wave; caller waits with s_wait_tensorcnt + barrier.
// This toolchain's builtin is the 6-arg form:
//   (uint32x4 g0, int32x8 g1, int32x4 g2, int32x4 g3, int32x8 pad, i32 cpol)
// ---------------------------------------------------------------------------
__device__ __forceinline__ void tdm_load_tile_2d(
    const float* gsrc, unsigned lds_byte_addr,
    unsigned tensor_d0, unsigned tensor_d1,
    unsigned tile_d0, unsigned tile_d1, unsigned d0_stride)
{
  unsigned long long ga = (unsigned long long)gsrc;
  // group0: [1:0]=count=1, [63:32]=lds_addr, [120:64]=global_addr, [127:126]=type 2
  u32x4 g0 = { 1u,
               lds_byte_addr,
               (unsigned)(ga & 0xffffffffull),
               (unsigned)((ga >> 32) & 0x01ffffffull) | (2u << 30) };
  // group1: [17:16]=data_size(2 -> 4B), [79:48]=tensor_dim0, [111:80]=tensor_dim1,
  //         [127:112]=tile_dim0, [143:128]=tile_dim1, [207:160]=tensor_dim0_stride
  i32x8 g1 = { (int)(2u << 16),
               (int)((tensor_d0 & 0xffffu) << 16),
               (int)(((tensor_d1 & 0xffffu) << 16) | ((tensor_d0 >> 16) & 0xffffu)),
               (int)(((tile_d0 & 0xffffu) << 16) | ((tensor_d1 >> 16) & 0xffffu)),
               (int)(tile_d1 & 0xffffu),
               (int)d0_stride,
               0, 0 };
  i32x4 z4 = { 0, 0, 0, 0 };          // groups 2/3 unused (<=2D tensor)
  i32x8 z8 = { 0, 0, 0, 0, 0, 0, 0, 0 };
  __builtin_amdgcn_tensor_load_to_lds(g0, g1, z4, z4, z8, 0);
}

// ---------------------------------------------------------------------------
// K1: X1[h] = x @ w1[h], X2[h] = x @ w2[h]   (8 GEMMs, M=1024 N=128 K=128)
// 4 waves / block, 1 tile / wave. 8 batch * 64 mt * 8 nt = 4096 tiles.
// ---------------------------------------------------------------------------
__global__ void k1_xw(const float* __restrict__ x,
                      const float* __restrict__ w1,
                      const float* __restrict__ w2,
                      float* __restrict__ X1, float* __restrict__ X2)
{
  const int lane = threadIdx.x & 31;
  const int t  = blockIdx.x * 4 + (threadIdx.x >> 5);
  const int b  = t >> 9;           // batch 0..7
  const int mt = (t >> 3) & 63;
  const int nt = t & 7;
  const int h  = b >> 1;
  const float* W = ((b & 1) ? w2 : w1) + h * DIM * DIM;
  float*       C = ((b & 1) ? X2 : X1) + (size_t)h * N_NODES * DIM;
  v8f c = wmma_tile_f16(x + mt * 16 * DIM, DIM, 1,
                        W + nt * 16,       DIM, 1, DIM, lane);
  store_tile<false>(C + mt * 16 * DIM + nt * 16, DIM, lane, c);
}

// ---------------------------------------------------------------------------
// K2: attention scores (the VALU-bound core, kept in f32).
//   Pun[h,i,j] = exp(8*tanh(s/8)) * A_shape[i,j],  RS[h,i] = sum_j Pun
// Block = 256 threads = 4 rows x 64 cols.  x2 tiles (64x128 f32 = 32KB) are
// staged into LDS by the Tensor Data Mover (one issuing wave, TENSORcnt wait,
// workgroup barrier releases the other 7 waves).
// ---------------------------------------------------------------------------
__global__ void k2_scores(const float* __restrict__ X1,
                          const float* __restrict__ X2,
                          const float* __restrict__ avec,
                          const float* __restrict__ A_shape,
                          float* __restrict__ Pun,
                          float* __restrict__ RS)
{
  __shared__ float sx1[4 * DIM];
  __shared__ float sa[DIM];
  __shared__ float sx2[64 * DIM];
  __shared__ float red[256];

  const int tid = threadIdx.x;
  const int h   = blockIdx.x >> 8;     // 4 heads
  const int ib  = blockIdx.x & 255;    // 256 row-blocks of 4
  const int ti  = tid >> 6;            // 0..3 row within block
  const int tj  = tid & 63;            // 0..63 col within tile
  const int i   = ib * 4 + ti;

  const float* X1h = X1 + (size_t)h * N_NODES * DIM;
  const float* X2h = X2 + (size_t)h * N_NODES * DIM;

  for (int idx = tid; idx < 4 * DIM; idx += 256)
    sx1[idx] = X1h[(size_t)(ib * 4) * DIM + idx];
  if (tid < DIM) sa[tid] = avec[h * DIM + tid];

  // LDS byte address of sx2 = low 32 bits of its flat (shared-aperture) addr
  const unsigned sx2_lds = (unsigned)(unsigned long long)(void*)sx2;

  const float* Arow = A_shape + (size_t)i * N_NODES;
  float*       Prow = Pun + ((size_t)h * N_NODES + i) * N_NODES;

  float rsum = 0.f;
  for (int jt = 0; jt < N_NODES; jt += 64) {
    __syncthreads();   // previous tile fully consumed (and sx1/sa on first pass)
    if (tid < 32) {    // one wave drives the TDM
      tdm_load_tile_2d(X2h + (size_t)jt * DIM, sx2_lds,
                       /*tensor_d0=*/DIM, /*tensor_d1=*/N_NODES,
                       /*tile_d0=*/DIM, /*tile_d1=*/64, /*d0_stride=*/DIM);
      __builtin_amdgcn_s_wait_tensorcnt(0);
    }
    if (jt + 64 < N_NODES)  // hint next tile toward L2 (global_prefetch_b8)
      __builtin_prefetch(&X2h[(size_t)(jt + 64) * DIM + tid], 0, 0);
    __syncthreads();   // tile visible to all 8 waves

    const float* r1 = &sx1[ti * DIM];
    const float* r2 = &sx2[tj * DIM];
    float acc = 0.f;
#pragma unroll 8
    for (int k = 0; k < DIM; ++k) {
      float v = r1[k] + r2[k];
      acc += sa[k] * fmaxf(v, 0.01f * v);   // leaky_relu(v, 0.01)
    }
    const float s = tanhf(acc * 0.125f) * 8.f;
    const int   j = jt + tj;
    const float e = __expf(s) * Arow[j];
    Prow[j] = e;
    rsum += e;
  }

  red[tid] = rsum;
  __syncthreads();
  for (int sft = 32; sft > 0; sft >>= 1) {
    if (tj < sft) red[tid] += red[tid + sft];
    __syncthreads();
  }
  if (tj == 0) RS[h * N_NODES + i] = red[ti * 64];
}

// ---------------------------------------------------------------------------
// K3: AGG[h] = (Pun[h] @ x) / RS[h][row]  (softmax normalization fused into
// the WMMA epilogue).  M=1024 N=128 K=1024 per head.
// ---------------------------------------------------------------------------
__global__ void k3_agg(const float* __restrict__ Pun,
                       const float* __restrict__ x,
                       const float* __restrict__ RS,
                       float* __restrict__ AGG)
{
  const int lane = threadIdx.x & 31;
  const int t  = blockIdx.x * 4 + (threadIdx.x >> 5);
  const int h  = t >> 9;
  const int mt = (t >> 3) & 63;
  const int nt = t & 7;
  const float* Ap = Pun + (size_t)h * N_NODES * N_NODES + (size_t)mt * 16 * N_NODES;
  v8f c = wmma_tile_f16(Ap, N_NODES, 1, x + nt * 16, DIM, 1, N_NODES, lane);

  const float* rs = RS + h * N_NODES + mt * 16;
  float* C = AGG + (size_t)h * N_NODES * DIM + mt * 16 * DIM + nt * 16;
  const int hl = lane >> 4, n = lane & 15;
#pragma unroll
  for (int v = 0; v < 8; ++v) {
    int m = v + hl * 8;
    C[m * DIM + n] = c[v] / rs[m];
  }
}

// ---------------------------------------------------------------------------
// K4: HC[:, h*128:(h+1)*128] = leaky_relu(AGG[h] @ Wlin[h]^T)
// Wlin is (out,in) applied as y @ W^T, so B(k,n) = Wlin[n*DIM + k]: the per-
// lane B column gather is a *contiguous* row of Wlin.
// ---------------------------------------------------------------------------
__global__ void k4_proj(const float* __restrict__ AGG,
                        const float* __restrict__ Wlin,
                        float* __restrict__ HC)
{
  const int lane = threadIdx.x & 31;
  const int t  = blockIdx.x * 4 + (threadIdx.x >> 5);
  const int h  = t >> 9;
  const int mt = (t >> 3) & 63;
  const int nt = t & 7;
  const float* Ap = AGG + (size_t)h * N_NODES * DIM + mt * 16 * DIM;
  const float* Bp = Wlin + h * DIM * DIM + nt * 16 * DIM;  // B(k,n)=Bp[n*DIM+k]
  v8f c = wmma_tile_f16(Ap, DIM, 1, Bp, 1, DIM, DIM, lane);
  float* C = HC + (size_t)mt * 16 * (HEADS * DIM) + h * DIM + nt * 16;
  store_tile<true>(C, HEADS * DIM, lane, c);
}

// ---------------------------------------------------------------------------
// K5: OUT = HC @ w   (M=1024 N=128 K=512)
// ---------------------------------------------------------------------------
__global__ void k5_out(const float* __restrict__ HC,
                       const float* __restrict__ w,
                       float* __restrict__ OUT)
{
  const int lane = threadIdx.x & 31;
  const int t  = blockIdx.x * 4 + (threadIdx.x >> 5);
  const int mt = t >> 3;
  const int nt = t & 7;
  const float* Ap = HC + (size_t)mt * 16 * (HEADS * DIM);
  v8f c = wmma_tile_f16(Ap, HEADS * DIM, 1, w + nt * 16, DIM, 1,
                        HEADS * DIM, lane);
  store_tile<false>(OUT + mt * 16 * DIM + nt * 16, DIM, lane, c);
}

// ---------------------------------------------------------------------------
extern "C" void kernel_launch(void* const* d_in, const int* in_sizes, int n_in,
                              void* d_out, int out_size, void* d_ws, size_t ws_size,
                              hipStream_t stream)
{
  (void)in_sizes; (void)n_in; (void)out_size; (void)ws_size;
  const float* x       = (const float*)d_in[0];
  const float* A_shape = (const float*)d_in[1];
  const float* w1      = (const float*)d_in[2];
  const float* w2      = (const float*)d_in[3];
  const float* avec    = (const float*)d_in[4];
  const float* Wlin    = (const float*)d_in[5];
  const float* w       = (const float*)d_in[6];

  float* ws  = (float*)d_ws;
  float* X1  = ws;                                        // 4*1024*128
  float* X2  = X1  + (size_t)HEADS * N_NODES * DIM;       // 4*1024*128
  float* Pun = X2  + (size_t)HEADS * N_NODES * DIM;       // 4*1024*1024
  float* RS  = Pun + (size_t)HEADS * N_NODES * N_NODES;   // 4*1024
  float* AGG = RS  + (size_t)HEADS * N_NODES;             // 4*1024*128
  float* HC  = AGG + (size_t)HEADS * N_NODES * DIM;       // 1024*512

  k1_xw    <<<1024, 128, 0, stream>>>(x, w1, w2, X1, X2);
  k2_scores<<<1024, 256, 0, stream>>>(X1, X2, avec, A_shape, Pun, RS);
  k3_agg   <<< 512, 128, 0, stream>>>(Pun, x, RS, AGG);
  k4_proj  <<< 512, 128, 0, stream>>>(AGG, Wlin, HC);
  k5_out   <<< 128, 128, 0, stream>>>(HC, w, (float*)d_out);
}
